// TemporalAttention_53128745452043
// MI455X (gfx1250) — matrix operand order repaired and checked by reading, not compiled, not run on hardware
//
#include <hip/hip_runtime.h>

// ---------------------------------------------------------------------------
// Causal attention with QKV projections for MI455X (gfx1250).
// All GEMMs: bf16 WMMA, async global->LDS staged 128x128 block tiles,
// 32x64 wave tiles (8 WMMA per 6 fragment loads). f32->bf16 conversion is
// hoisted out of all GEMM hot loops.
// B=4, S=4096, H=512.  d_out = [context (B*S*H f32)] ++ [attn (B*S*S f32)].
// ---------------------------------------------------------------------------

#define B_ 4
#define S_ 4096
#define H_ 512
#define NEG_INF_ (-10000000.0f)
#define SCALE_ 0.04419417382415922f  // 1/sqrt(512)
#define KT_ 64                       // K-tile staged in LDS
#define ALD_ 72                      // padded LDS row stride (bf16 elems)

typedef __attribute__((ext_vector_type(8)))  __bf16   v8bf;
typedef __attribute__((ext_vector_type(16))) __bf16   v16bf;
typedef __attribute__((ext_vector_type(8)))  float    v8f;
typedef __attribute__((ext_vector_type(4)))  float    v4f;
typedef __attribute__((ext_vector_type(8)))  unsigned v8u;
typedef __attribute__((ext_vector_type(4)))  unsigned v4u;
typedef __attribute__((ext_vector_type(4)))  int      v4i;

// ---- async global->LDS copy (signature verified via hipcc diagnostic) ------
#if defined(__has_builtin)
#if __has_builtin(__builtin_amdgcn_global_load_async_to_lds_b128)
#define HAVE_ASYNC_LDS 1
#endif
#endif

#if defined(HAVE_ASYNC_LDS)
#define COPY16(dst, src)                                                   \
  __builtin_amdgcn_global_load_async_to_lds_b128(                          \
      (__attribute__((address_space(1))) v4i*)(src),                       \
      (__attribute__((address_space(3))) v4i*)(dst), 0, 0)
#else
#define COPY16(dst, src) (*(v8bf*)(dst) = *(const v8bf*)(src))
#endif

__device__ __forceinline__ void wait_async_copies() {
#if defined(HAVE_ASYNC_LDS)
#if __has_builtin(__builtin_amdgcn_s_wait_asynccnt)
  __builtin_amdgcn_s_wait_asynccnt(0);
#else
  asm volatile("s_wait_asynccnt 0" ::: "memory");
#endif
#endif
}

// ---- conversions -----------------------------------------------------------
// two f32 -> one dword {lo: bf16(a), hi: bf16(b)}; 3 VALU (add, add, perm)
__device__ __forceinline__ unsigned pack2_bf16(float a, float b) {
  unsigned ua = __builtin_bit_cast(unsigned, a) + 0x8000u;  // round-half-up
  unsigned ub = __builtin_bit_cast(unsigned, b) + 0x8000u;
  return __builtin_amdgcn_perm(ub, ua, 0x07060302u);
}

// single f32 -> bf16, round-half-up (2 VALU)
__device__ __forceinline__ __bf16 cvt_bf16(float f) {
  unsigned r = (__builtin_bit_cast(unsigned, f) + 0x8000u) >> 16;
  return __builtin_bit_cast(__bf16, (unsigned short)r);
}

__device__ __forceinline__ v16bf concat8(v8bf lo, v8bf hi) {
  return __builtin_shufflevector(lo, hi, 0, 1, 2, 3, 4, 5, 6, 7,
                                 8, 9, 10, 11, 12, 13, 14, 15);
}

__device__ __forceinline__ v8f wmma_bf16(v16bf a, v16bf b, v8f c) {
  return __builtin_amdgcn_wmma_f32_16x16x32_bf16(
      false, a, false, b, (short)0, c, false, false);
}

// ---- fragment builders (bf16 sources only; no conversion in GEMM loops) ----
// A fragment (16x32 MxK): lanes 0-15 hold M=lane, K=[k0..k0+7]+[k0+16..k0+23];
// lanes 16-31 hold M=lane-16, K=[k0+8..15]+[k0+24..31].
__device__ __forceinline__ v16bf frag_a_bf16(const __bf16* __restrict__ A, int lda,
                                             int row0, int k0, int lane) {
  const int m = lane & 15, hh = lane >> 4;
  const __bf16* p = A + (size_t)(row0 + m) * lda + k0 + hh * 8;
  return concat8(*(const v8bf*)(p), *(const v8bf*)(p + 16));
}

// B fragment (32x16 KxN), B[k,n] = W[n,k]: lanes 0-15 N=lane K=[k0..k0+7]+[k0+16..k0+23];
// lanes 16-31 N=lane-16 K=[k0+8..15]+[k0+24..31].
__device__ __forceinline__ v16bf frag_b_bf16(const __bf16* __restrict__ W, int ldw,
                                             int col0, int k0, int lane) {
  const int n = lane & 15, hh = lane >> 4;
  const __bf16* p = W + (size_t)(col0 + n) * ldw + k0 + hh * 8;
  return concat8(*(const v8bf*)(p), *(const v8bf*)(p + 16));
}

// ---------------------------------------------------------------------------
// Kernel 0: elementwise f32 -> bf16 (8 elems/thread, packed dword stores)
// ---------------------------------------------------------------------------
__global__ __launch_bounds__(256)
void cvt_kernel(const float* __restrict__ in, __bf16* __restrict__ out, int n8) {
  const int i = blockIdx.x * 256 + threadIdx.x;
  if (i >= n8) return;
  const float* p = in + (size_t)i * 8;
  v4f x0 = *(const v4f*)(p);
  v4f x1 = *(const v4f*)(p + 4);
  v4u u;
  u[0] = pack2_bf16(x0[0], x0[1]);
  u[1] = pack2_bf16(x0[2], x0[3]);
  u[2] = pack2_bf16(x1[0], x1[1]);
  u[3] = pack2_bf16(x1[2], x1[3]);
  *(v4u*)(out + (size_t)i * 8) = u;
}

// ---------------------------------------------------------------------------
// Kernel 1: projection GEMM. out[s,o] = sum_h X[s,h]*W[o,h] + bias[o].
// X,W pre-converted bf16. LDS-staged like the attention GEMMs.
// vmode=0: out row-major [B*S,H]; vmode=1: out = vT[b][h][s] (transposed).
// ---------------------------------------------------------------------------
__global__ __launch_bounds__(256)
void proj_kernel(const __bf16* __restrict__ X, const __bf16* __restrict__ W,
                 const float* __restrict__ bias, __bf16* __restrict__ out,
                 int vmode) {
  const int t    = threadIdx.x;
  const int lane = t & 31;
  const int wave = t >> 5;
  const int rblk = blockIdx.y * 128;  // rows within B*S
  const int cblk = blockIdx.x * 128;  // output features

  const __bf16* Xp = X + (size_t)rblk * H_;
  const __bf16* Wp = W + (size_t)cblk * H_;

  __shared__ __bf16 sX[128][ALD_];  // 18 KB
  __shared__ __bf16 sW[128][ALD_];  // 18 KB

  const int row0 = (wave & 3) * 32;
  const int col0 = (wave >> 2) * 64;

  v8f acc[2][4] = {};
  for (int k0 = 0; k0 < H_; k0 += KT_) {
#pragma unroll
    for (int i = 0; i < 4; ++i) {
      const int c = t + 256 * i;
      const int r = c >> 3, ko = (c & 7) * 8;
      COPY16(&sX[r][ko], Xp + (size_t)r * H_ + k0 + ko);
      COPY16(&sW[r][ko], Wp + (size_t)r * H_ + k0 + ko);
    }
    wait_async_copies();
    __syncthreads();

#pragma unroll
    for (int kk = 0; kk < KT_; kk += 32) {
      v16bf a[2], bb[4];
      a[0] = frag_a_bf16(&sX[0][0], ALD_, row0, kk, lane);
      a[1] = frag_a_bf16(&sX[0][0], ALD_, row0 + 16, kk, lane);
#pragma unroll
      for (int j = 0; j < 4; ++j)
        bb[j] = frag_b_bf16(&sW[0][0], ALD_, col0 + 16 * j, kk, lane);
#pragma unroll
      for (int i = 0; i < 2; ++i)
#pragma unroll
        for (int j = 0; j < 4; ++j)
          acc[i][j] = wmma_bf16(a[i], bb[j], acc[i][j]);
    }
    __syncthreads();
  }

  const int n  = lane & 15;
  const int mb = (lane >> 4) * 8;  // C/D layout: VGPR r -> row r + 8*(lane>=16)
  float bs[4];
#pragma unroll
  for (int j = 0; j < 4; ++j) bs[j] = bias[cblk + col0 + 16 * j + n];

  if (vmode == 0) {
#pragma unroll
    for (int i = 0; i < 2; ++i)
#pragma unroll
      for (int j = 0; j < 4; ++j)
#pragma unroll
        for (int r = 0; r < 8; ++r) {
          const int m = rblk + row0 + 16 * i + mb + r;
          out[(size_t)m * H_ + cblk + col0 + 16 * j + n] =
              cvt_bf16(acc[i][j][r] + bs[j]);
        }
  } else {
    // transposed store: vT[b][h][s], m = b*S + s (tiles never cross batches)
#pragma unroll
    for (int i = 0; i < 2; ++i)
#pragma unroll
      for (int j = 0; j < 4; ++j)
#pragma unroll
        for (int r = 0; r < 8; ++r) {
          const int m = rblk + row0 + 16 * i + mb + r;
          const size_t base = (size_t)(m >> 12) * H_ * S_;
          const int s = m & (S_ - 1);
          const int h = cblk + col0 + 16 * j + n;
          out[base + (size_t)h * S_ + s] = cvt_bf16(acc[i][j][r] + bs[j]);
        }
  }
}

// ---------------------------------------------------------------------------
// Kernel 2: scores = scale * Q K^T. 128x128 block tile staged in LDS via
// async copies; skips tiles fully above the causal diagonal.
// ---------------------------------------------------------------------------
__global__ __launch_bounds__(256)
void scores_kernel(const __bf16* __restrict__ qbf, const __bf16* __restrict__ kbf,
                   float* __restrict__ attn) {
  const int b    = blockIdx.z;
  const int qblk = blockIdx.y * 128;
  const int kblk = blockIdx.x * 128;
  if (kblk > qblk + 127) return;  // entire tile masked by causality

  const __bf16* q = qbf + (size_t)b * S_ * H_ + (size_t)qblk * H_;
  const __bf16* k = kbf + (size_t)b * S_ * H_ + (size_t)kblk * H_;
  float* out = attn + (size_t)b * S_ * S_;

  __shared__ __bf16 sQ[128][ALD_];  // 18 KB
  __shared__ __bf16 sK[128][ALD_];  // 18 KB

  const int t    = threadIdx.x;
  const int lane = t & 31;
  const int wave = t >> 5;
  const int row0 = (wave & 3) * 32;   // local q-row base
  const int col0 = (wave >> 2) * 64;  // local k-row base

  v8f acc[2][4] = {};
  for (int k0 = 0; k0 < H_; k0 += KT_) {
#pragma unroll
    for (int i = 0; i < 4; ++i) {
      const int c = t + 256 * i;
      const int r = c >> 3, ko = (c & 7) * 8;
      COPY16(&sQ[r][ko], q + (size_t)r * H_ + k0 + ko);
      COPY16(&sK[r][ko], k + (size_t)r * H_ + k0 + ko);
    }
    wait_async_copies();
    __syncthreads();

#pragma unroll
    for (int kk = 0; kk < KT_; kk += 32) {
      v16bf a[2], bb[4];
      a[0] = frag_a_bf16(&sQ[0][0], ALD_, row0, kk, lane);
      a[1] = frag_a_bf16(&sQ[0][0], ALD_, row0 + 16, kk, lane);
#pragma unroll
      for (int j = 0; j < 4; ++j)
        bb[j] = frag_b_bf16(&sK[0][0], ALD_, col0 + 16 * j, kk, lane);
#pragma unroll
      for (int i = 0; i < 2; ++i)
#pragma unroll
        for (int j = 0; j < 4; ++j)
          acc[i][j] = wmma_bf16(a[i], bb[j], acc[i][j]);
    }
    __syncthreads();
  }

  const int n = lane & 15, mb = (lane >> 4) * 8;
  const int gr = qblk + row0;
  const int gc = kblk + col0;
#pragma unroll
  for (int i = 0; i < 2; ++i)
#pragma unroll
    for (int j = 0; j < 4; ++j)
#pragma unroll
      for (int r = 0; r < 8; ++r) {
        const int m = gr + 16 * i + mb + r;
        out[(size_t)m * S_ + gc + 16 * j + n] = acc[i][j][r] * SCALE_;
      }
}

// ---------------------------------------------------------------------------
// Kernel 3: causal softmax over one S-length row per block (256 thr, 16/thr).
// Writes normalized f32 attn to d_out AND a packed bf16 copy to workspace.
// ---------------------------------------------------------------------------
__global__ __launch_bounds__(256)
void softmax_kernel(float* __restrict__ attn, __bf16* __restrict__ attn_bf) {
  const int row = blockIdx.x;        // 0 .. B*S-1
  const int qi  = row & (S_ - 1);    // query index within batch
  float* p = attn + (size_t)row * S_;
  const int t = threadIdx.x, lane = t & 31, wave = t >> 5;
  const int base = t * 16;

  float v[16];
#pragma unroll
  for (int j = 0; j < 4; ++j) {
    v4f x = *(const v4f*)(p + base + 4 * j);
#pragma unroll
    for (int i = 0; i < 4; ++i) v[4 * j + i] = x[i];
  }

  float mx = -3.0e38f;
#pragma unroll
  for (int i = 0; i < 16; ++i) {
    const float s = (base + i <= qi) ? v[i] : NEG_INF_;
    v[i] = s;
    mx = fmaxf(mx, s);
  }

  __shared__ float red[8];
#pragma unroll
  for (int o = 16; o > 0; o >>= 1) mx = fmaxf(mx, __shfl_down(mx, o, 32));
  if (lane == 0) red[wave] = mx;
  __syncthreads();
  if (t == 0) {
    float m = red[0];
#pragma unroll
    for (int i = 1; i < 8; ++i) m = fmaxf(m, red[i]);
    red[0] = m;
  }
  __syncthreads();
  mx = red[0];
  __syncthreads();  // red about to be reused

  float sum = 0.f;
#pragma unroll
  for (int i = 0; i < 16; ++i) {
    const float e = __expf(v[i] - mx);  // masked: exp(-1e7 - mx) -> 0
    v[i] = e;
    sum += e;
  }
#pragma unroll
  for (int o = 16; o > 0; o >>= 1) sum += __shfl_down(sum, o, 32);
  if (lane == 0) red[wave] = sum;
  __syncthreads();
  if (t == 0) {
    float s2 = 0.f;
#pragma unroll
    for (int i = 0; i < 8; ++i) s2 += red[i];
    red[0] = s2;
  }
  __syncthreads();
  const float inv = 1.0f / red[0];

#pragma unroll
  for (int i = 0; i < 16; ++i) v[i] *= inv;

#pragma unroll
  for (int j = 0; j < 4; ++j) {
    v4f x;
#pragma unroll
    for (int i = 0; i < 4; ++i) x[i] = v[4 * j + i];
    *(v4f*)(p + base + 4 * j) = x;
  }

  // bf16 copy for the context GEMM (32B per thread, contiguous)
  v8u pk;
#pragma unroll
  for (int j = 0; j < 8; ++j) pk[j] = pack2_bf16(v[2 * j], v[2 * j + 1]);
  *(v8u*)(attn_bf + (size_t)row * S_ + base) = pk;
}

// ---------------------------------------------------------------------------
// Kernel 4: context = attn_bf @ V via vT (staged in LDS). Block tile 128x128,
// wave tile 32x64. K-loop truncated at the causal diagonal.
// ---------------------------------------------------------------------------
__global__ __launch_bounds__(256)
void context_kernel(const __bf16* __restrict__ attn_bf,
                    const __bf16* __restrict__ vT, float* __restrict__ ctx) {
  const int b = blockIdx.z;
  const __bf16* A  = attn_bf + (size_t)b * S_ * S_;
  const __bf16* Bt = vT + (size_t)b * H_ * S_;
  float* out = ctx + (size_t)b * S_ * H_;

  __shared__ __bf16 sV[128][ALD_];  // 18 KB (128 h-rows x 64 k)

  const int t    = threadIdx.x;
  const int lane = t & 31;
  const int wave = t >> 5;
  const int qblk = blockIdx.y * 128;
  const int hblk = blockIdx.x * 128;
  const int row0 = qblk + (wave & 3) * 32;
  const int col0 = (wave >> 2) * 64;  // local h base inside 128-wide tile

  const int kmax = qblk + 128;  // attn rows in this tile are 0 beyond this

  v8f acc[2][4] = {};
  for (int k0 = 0; k0 < kmax; k0 += KT_) {
#pragma unroll
    for (int i = 0; i < 4; ++i) {
      const int c = t + 256 * i;
      const int r = c >> 3, ko = (c & 7) * 8;
      COPY16(&sV[r][ko], Bt + (size_t)(hblk + r) * S_ + k0 + ko);
    }
    if (k0 + KT_ < kmax)
      __builtin_prefetch(A + (size_t)row0 * S_ + k0 + KT_, 0, 1);
    wait_async_copies();
    __syncthreads();

#pragma unroll
    for (int kk = 0; kk < KT_; kk += 32) {
      v16bf a[2], bb[4];
      a[0] = frag_a_bf16(A, S_, row0, k0 + kk, lane);
      a[1] = frag_a_bf16(A, S_, row0 + 16, k0 + kk, lane);
#pragma unroll
      for (int j = 0; j < 4; ++j)
        bb[j] = frag_b_bf16(&sV[0][0], ALD_, col0 + 16 * j, kk, lane);
#pragma unroll
      for (int i = 0; i < 2; ++i)
#pragma unroll
        for (int j = 0; j < 4; ++j)
          acc[i][j] = wmma_bf16(a[i], bb[j], acc[i][j]);
    }
    __syncthreads();
  }

  const int n = lane & 15, mb = (lane >> 4) * 8;
  const int gc = hblk + col0;
#pragma unroll
  for (int i = 0; i < 2; ++i)
#pragma unroll
    for (int j = 0; j < 4; ++j)
#pragma unroll
      for (int r = 0; r < 8; ++r) {
        const int m = row0 + 16 * i + mb + r;
        out[(size_t)m * H_ + gc + 16 * j + n] = acc[i][j][r];
      }
}

// ---------------------------------------------------------------------------
extern "C" void kernel_launch(void* const* d_in, const int* in_sizes, int n_in,
                              void* d_out, int out_size, void* d_ws, size_t ws_size,
                              hipStream_t stream) {
  (void)in_sizes; (void)n_in; (void)out_size; (void)ws_size;

  const float* Xq = (const float*)d_in[0];
  const float* Xk = (const float*)d_in[1];
  const float* Xv = (const float*)d_in[2];
  const float* Wq = (const float*)d_in[3];
  const float* bq = (const float*)d_in[4];
  const float* Wk = (const float*)d_in[5];
  const float* bk = (const float*)d_in[6];
  const float* Wv = (const float*)d_in[7];
  const float* bv = (const float*)d_in[8];

  float* ctx  = (float*)d_out;               // B*S*H f32
  float* attn = ctx + (size_t)B_ * S_ * H_;  // B*S*S f32

  const size_t BSH = (size_t)B_ * S_ * H_;   // 8388608
  const size_t HH  = (size_t)H_ * H_;        // 262144
  __bf16* qbf     = (__bf16*)d_ws;           // [B*S, H]   16 MB
  __bf16* kbf     = qbf + BSH;               // [B*S, H]   16 MB
  __bf16* vT      = kbf + BSH;               // [B][H][S]  16 MB
  __bf16* attn_bf = vT + BSH;                // [B][S][S] 128 MB
  __bf16* Xqb     = attn_bf + (size_t)B_ * S_ * S_;  // 16 MB each
  __bf16* Xkb     = Xqb + BSH;
  __bf16* Xvb     = Xkb + BSH;
  __bf16* Wqb     = Xvb + BSH;               // 0.5 MB each
  __bf16* Wkb     = Wqb + HH;
  __bf16* Wvb     = Wkb + HH;

  dim3 blk(256);
  const int gX = (int)(BSH / 8 / 256);  // 4096 blocks
  const int gW = (int)(HH / 8 / 256);   // 128 blocks
  cvt_kernel<<<gX, blk, 0, stream>>>(Xq, Xqb, (int)(BSH / 8));
  cvt_kernel<<<gX, blk, 0, stream>>>(Xk, Xkb, (int)(BSH / 8));
  cvt_kernel<<<gX, blk, 0, stream>>>(Xv, Xvb, (int)(BSH / 8));
  cvt_kernel<<<gW, blk, 0, stream>>>(Wq, Wqb, (int)(HH / 8));
  cvt_kernel<<<gW, blk, 0, stream>>>(Wk, Wkb, (int)(HH / 8));
  cvt_kernel<<<gW, blk, 0, stream>>>(Wv, Wvb, (int)(HH / 8));

  dim3 pgrid(H_ / 128, (B_ * S_) / 128);  // 4 x 128
  proj_kernel<<<pgrid, blk, 0, stream>>>(Xqb, Wqb, bq, qbf, 0);
  proj_kernel<<<pgrid, blk, 0, stream>>>(Xkb, Wkb, bk, kbf, 0);
  proj_kernel<<<pgrid, blk, 0, stream>>>(Xvb, Wvb, bv, vT, 1);

  scores_kernel<<<dim3(S_ / 128, S_ / 128, B_), blk, 0, stream>>>(qbf, kbf, attn);
  softmax_kernel<<<dim3(B_ * S_), blk, 0, stream>>>(attn, attn_bf);
  context_kernel<<<dim3(H_ / 128, S_ / 128, B_), blk, 0, stream>>>(attn_bf, vT, ctx);
}
